// SimpleTernaryNet_16363825397997
// MI455X (gfx1250) — matrix operand order repaired and checked by reading, not compiled, not run on hardware
//
#include <hip/hip_runtime.h>
#include <math.h>

typedef __attribute__((ext_vector_type(16))) _Float16 v16h;
typedef __attribute__((ext_vector_type(8)))  float    v8f;
typedef __attribute__((ext_vector_type(4)))  int      v4i;

#define EPSV 1e-5f

// ---------------------------------------------------------------------------
// Ternarize (TWN): delta = 0.7*mean|w|; alpha = sum(|w|*mask)/max(sum mask,1)
// Writes quantized f16 weights into a zero-padded [rows_out, cols_out] buffer.
// ---------------------------------------------------------------------------
__global__ __launch_bounds__(256)
void ternarize_kernel(const float* __restrict__ w, int rows, int cols,
                      _Float16* __restrict__ out, int rows_out, int cols_out) {
  __shared__ float red[256];
  __shared__ float s_delta, s_tmp, s_alpha;
  const int tid = threadIdx.x;
  const int n = rows * cols;

  for (int i = tid; i < rows_out * cols_out; i += 256) out[i] = (_Float16)0.f;

  float s = 0.f;
  for (int i = tid; i < n; i += 256) s += fabsf(w[i]);
  red[tid] = s; __syncthreads();
  for (int off = 128; off > 0; off >>= 1) {
    if (tid < off) red[tid] += red[tid + off];
    __syncthreads();
  }
  if (tid == 0) s_delta = 0.7f * red[0] / (float)n;
  __syncthreads();
  const float delta = s_delta;

  float sa = 0.f, cnt = 0.f;
  for (int i = tid; i < n; i += 256) {
    float a = fabsf(w[i]);
    if (a > delta) { sa += a; cnt += 1.f; }
  }
  red[tid] = sa; __syncthreads();
  for (int off = 128; off > 0; off >>= 1) {
    if (tid < off) red[tid] += red[tid + off];
    __syncthreads();
  }
  if (tid == 0) s_tmp = red[0];
  __syncthreads();
  red[tid] = cnt; __syncthreads();
  for (int off = 128; off > 0; off >>= 1) {
    if (tid < off) red[tid] += red[tid + off];
    __syncthreads();
  }
  if (tid == 0) s_alpha = s_tmp / fmaxf(red[0], 1.0f);
  __syncthreads();
  const float alpha = s_alpha;

  for (int i = tid; i < n; i += 256) {
    float wv = w[i];
    float q = (fabsf(wv) > delta) ? (wv > 0.f ? alpha : -alpha) : 0.f;
    int r = i / cols, c = i - r * cols;
    out[r * cols_out + c] = (_Float16)q;
  }
}

// ---------------------------------------------------------------------------
// Fragment helpers: lane layouts used by this code (wave32).
// A (16x32 f16, MxK): lane m=lane&15; K=(lane>>4)*16+e (contiguous 16)
// B (32x16 f16, KxN): lane n=lane&15; K=(lane>>4)*16+e (contiguous 16)
// D (16x16 f32):      lane col n=lane&15; elem i -> row m = i + 8*(lane>>4)
// ---------------------------------------------------------------------------
__device__ inline v16h load_a_frag(const _Float16* __restrict__ kchunk, int khalf) {
  v16h a;
  const _Float16* p0 = kchunk + khalf * 16;
  const _Float16* p1 = kchunk + khalf * 16 + 8;
#pragma unroll
  for (int e = 0; e < 8; ++e) { a[e] = p0[e]; a[e + 8] = p1[e]; }
  return a;
}

// ---------------------------------------------------------------------------
// conv1: [B,3,32,32] f32 -> WMMA (M=32ch, K=27->32) -> +bias,BN,ReLU -> f16
// One wave = 16 positions (half a row) x all 32 channels (2 WMMAs, B reused).
// ---------------------------------------------------------------------------
__global__ __launch_bounds__(256)
void conv1_wmma_kernel(const float* __restrict__ x,       // [B,3,32,32]
                       const _Float16* __restrict__ wq,   // [32,32] K-padded
                       const float* __restrict__ bias, const float* __restrict__ g,
                       const float* __restrict__ be, const float* __restrict__ mu,
                       const float* __restrict__ var,
                       _Float16* __restrict__ act)        // [B,32,32,32]
{
  const int wave = blockIdx.x * (blockDim.x >> 5) + (threadIdx.x >> 5);
  const int lane = threadIdx.x & 31;
  const int b = wave >> 6;
  const int ptile = wave & 63;
  const int p0 = ptile * 16;
  const int oy = p0 >> 5;
  const int ox0 = p0 & 31;
  const int n = lane & 15;
  const int khalf = lane >> 4;
  const int mrow = lane & 15;

  const float* xb = x + (size_t)b * 3 * 32 * 32;

  v16h bm;
#pragma unroll
  for (int e = 0; e < 16; ++e) {
    int k = khalf * 16 + e;
    float v = 0.f;
    if (k < 27) {
      int ci = k / 9, r = k - ci * 9;
      int iy = oy + r / 3 - 1;
      int ix = ox0 + n + (r % 3) - 1;
      if (iy >= 0 && iy < 32 && ix >= 0 && ix < 32)
        v = xb[ci * 1024 + iy * 32 + ix];
    }
    bm[e] = (_Float16)v;
  }

  v16h a0 = load_a_frag(wq + (size_t)mrow * 32, khalf);
  v16h a1 = load_a_frag(wq + (size_t)(mrow + 16) * 32, khalf);
  v8f c0 = {}, c1 = {};
  c0 = __builtin_amdgcn_wmma_f32_16x16x32_f16(false, a0, false, bm, (short)0, c0, false, false);
  c1 = __builtin_amdgcn_wmma_f32_16x16x32_f16(false, a1, false, bm, (short)0, c1, false, false);

#pragma unroll
  for (int i = 0; i < 8; ++i) {
    int mi = i + khalf * 8;
#pragma unroll
    for (int grp = 0; grp < 2; ++grp) {
      int c = mi + grp * 16;
      float inv = g[c] * rsqrtf(var[c] + EPSV);
      float val = ((grp ? c1[i] : c0[i]) + bias[c] - mu[c]) * inv + be[c];
      val = fmaxf(val, 0.f);
      act[((size_t)b * 32 + c) * 1024 + oy * 32 + ox0 + n] = (_Float16)val;
    }
  }
}

// ---------------------------------------------------------------------------
// 2x2 max pool over f16 NCHW.
// ---------------------------------------------------------------------------
__global__ __launch_bounds__(256)
void maxpool_kernel(const _Float16* __restrict__ in, _Float16* __restrict__ out,
                    int total, int H, int W) {
  int i = blockIdx.x * blockDim.x + threadIdx.x;
  if (i >= total) return;
  const int OW = W >> 1, OH = H >> 1;
  int ox = i % OW;
  int t = i / OW;
  int oy = t % OH;
  int bc = t / OH;
  const _Float16* p = in + ((size_t)bc * H + oy * 2) * W + ox * 2;
  float v = fmaxf(fmaxf((float)p[0], (float)p[1]),
                  fmaxf((float)p[W], (float)p[W + 1]));
  out[i] = (_Float16)v;
}

// ---------------------------------------------------------------------------
// conv2: pooled1 [B,32,16,16] f16 -> WMMA (M=64ch, K=288 = 9 steps) -> BN,ReLU
// Block (8 waves) = half an image (8 output rows): stage the whole 16 KB input image into LDS via
// the CDNA5 async Memory->LDS path (ASYNCcnt), then each wave computes one
// output row (16 positions) x all 64 channels with im2col reads from LDS.
// ---------------------------------------------------------------------------
__global__ __launch_bounds__(256)
void conv2_wmma_kernel(const _Float16* __restrict__ in,   // [B,32,16,16]
                       const _Float16* __restrict__ wq,   // [64,288]
                       const float* __restrict__ bias, const float* __restrict__ g,
                       const float* __restrict__ be, const float* __restrict__ mu,
                       const float* __restrict__ var,
                       _Float16* __restrict__ outp)       // [B,64,16,16]
{
  __shared__ _Float16 s_in[32 * 16 * 16];   // 16 KB = full input image

  const int tid = threadIdx.x;
  const int lane = tid & 31;
  const int b = blockIdx.x >> 1;
  const int half = blockIdx.x & 1;
  const int oy = half * 8 + (tid >> 5);     // one output row per wave
  const int n = lane & 15;
  const int khalf = lane >> 4;
  const int mrow = lane & 15;

  const _Float16* inb = in + (size_t)b * (32 * 16 * 16);

  // --- stage image: 256 threads x 64 B (4 x async B128 each) ---
#if defined(__gfx1250__) && __has_builtin(__builtin_amdgcn_global_load_async_to_lds_b128)
  {
    typedef __attribute__((address_space(1))) v4i gv4i_t;  // global int4
    typedef __attribute__((address_space(3))) v4i lv4i_t;  // LDS int4
    const char* gsrc = (const char*)inb + tid * 64;
    char* ldst = (char*)&s_in[0] + tid * 64;
#pragma unroll
    for (int j = 0; j < 4; ++j) {
      __builtin_amdgcn_global_load_async_to_lds_b128(
          (gv4i_t*)(gsrc + j * 16),
          (lv4i_t*)(ldst + j * 16),
          0, 0);
    }
#if __has_builtin(__builtin_amdgcn_s_wait_asynccnt)
    __builtin_amdgcn_s_wait_asynccnt(0);
#endif
  }
#else
  {
    const v4i* gsrc = (const v4i*)((const char*)inb + tid * 64);
    v4i* ldst = (v4i*)((char*)&s_in[0] + tid * 64);
#pragma unroll
    for (int j = 0; j < 4; ++j) ldst[j] = gsrc[j];
  }
#endif
  __syncthreads();

  v8f acc[4] = {};
  for (int kk = 0; kk < 9; ++kk) {
    const int kbase = kk * 32;
    v16h bm;
#pragma unroll
    for (int e = 0; e < 16; ++e) {
      int kg = kbase + khalf * 16 + e;       // < 288
      int ci = kg / 9, r = kg - ci * 9;
      int iy = oy + r / 3 - 1;
      int ix = n + (r % 3) - 1;
      float v = 0.f;
      if (iy >= 0 && iy < 16 && ix >= 0 && ix < 16)
        v = (float)s_in[(ci * 16 + iy) * 16 + ix];
      bm[e] = (_Float16)v;
    }
#pragma unroll
    for (int ct = 0; ct < 4; ++ct) {
      v16h a = load_a_frag(wq + (size_t)(ct * 16 + mrow) * 288 + kbase, khalf);
      acc[ct] = __builtin_amdgcn_wmma_f32_16x16x32_f16(false, a, false, bm, (short)0,
                                                       acc[ct], false, false);
    }
  }

#pragma unroll
  for (int ct = 0; ct < 4; ++ct) {
#pragma unroll
    for (int i = 0; i < 8; ++i) {
      int c = ct * 16 + i + khalf * 8;
      float inv = g[c] * rsqrtf(var[c] + EPSV);
      float val = ((float)acc[ct][i] + bias[c] - mu[c]) * inv + be[c];
      val = fmaxf(val, 0.f);
      outp[((size_t)b * 64 + c) * 256 + oy * 16 + n] = (_Float16)val;
    }
  }
}

// ---------------------------------------------------------------------------
// fc1: Y[1024,256] = relu(A[1024,4096] @ Wq^T + fb1), f16 in, f16 out.
// ---------------------------------------------------------------------------
__global__ __launch_bounds__(256)
void fc1_wmma_kernel(const _Float16* __restrict__ actv,  // [1024,4096]
                     const _Float16* __restrict__ wq,    // [256,4096]
                     const float* __restrict__ bias,     // [256]
                     _Float16* __restrict__ y)           // [1024,256]
{
  const int wave = blockIdx.x * (blockDim.x >> 5) + (threadIdx.x >> 5);
  const int lane = threadIdx.x & 31;
  const int mt = wave >> 4;
  const int nt = wave & 15;
  const int row0 = mt * 16, col0 = nt * 16;
  const int mrow = lane & 15, khalf = lane >> 4, ncol = lane & 15;

  const _Float16* arow = actv + (size_t)(row0 + mrow) * 4096;
  const _Float16* brow = wq + (size_t)(col0 + ncol) * 4096 + khalf * 16;

  v8f acc = {};
  for (int k0 = 0; k0 < 4096; k0 += 32) {
    __builtin_prefetch(brow + k0 + 256, 0, 1);   // global_prefetch_b8
    v16h a = load_a_frag(arow + k0, khalf);
    v16h b = *(const v16h*)(brow + k0);          // 32B contiguous, aligned
    acc = __builtin_amdgcn_wmma_f32_16x16x32_f16(false, a, false, b, (short)0,
                                                 acc, false, false);
  }
#pragma unroll
  for (int i = 0; i < 8; ++i) {
    int m = i + khalf * 8;
    int c = col0 + ncol;
    float v = fmaxf(acc[i] + bias[c], 0.f);
    y[(size_t)(row0 + m) * 256 + c] = (_Float16)v;
  }
}

// ---------------------------------------------------------------------------
// fc2: out[1024,10] f32 = Y1[1024,256] @ Wq^T + fb2 (N padded to 16).
// ---------------------------------------------------------------------------
__global__ __launch_bounds__(256)
void fc2_wmma_kernel(const _Float16* __restrict__ y1,    // [1024,256]
                     const _Float16* __restrict__ wq,    // [16,256] row-padded
                     const float* __restrict__ bias,     // [10]
                     float* __restrict__ out)            // [1024,10]
{
  const int wave = blockIdx.x * (blockDim.x >> 5) + (threadIdx.x >> 5);
  const int lane = threadIdx.x & 31;
  const int row0 = wave * 16;
  const int mrow = lane & 15, khalf = lane >> 4, ncol = lane & 15;

  const _Float16* arow = y1 + (size_t)(row0 + mrow) * 256;
  const _Float16* brow = wq + (size_t)ncol * 256 + khalf * 16;

  v8f acc = {};
#pragma unroll
  for (int k0 = 0; k0 < 256; k0 += 32) {
    v16h a = load_a_frag(arow + k0, khalf);
    v16h b = *(const v16h*)(brow + k0);
    acc = __builtin_amdgcn_wmma_f32_16x16x32_f16(false, a, false, b, (short)0,
                                                 acc, false, false);
  }
  if (ncol < 10) {
    float bi = bias[ncol];
#pragma unroll
    for (int i = 0; i < 8; ++i) {
      int m = i + khalf * 8;
      out[(size_t)(row0 + m) * 10 + ncol] = acc[i] + bi;
    }
  }
}

// ---------------------------------------------------------------------------
// Host side: ternarize -> conv1 -> pool -> conv2 -> pool -> fc1 -> fc2
// act2 aliases act1 (act1 dead after pool1); pooled2 aliases pooled1
// (dead after conv2). Peak ~87 MB.
// ---------------------------------------------------------------------------
extern "C" void kernel_launch(void* const* d_in, const int* in_sizes, int n_in,
                              void* d_out, int out_size, void* d_ws, size_t ws_size,
                              hipStream_t stream) {
  (void)in_sizes; (void)n_in; (void)out_size; (void)ws_size;
  const float* x   = (const float*)d_in[0];
  const float* w1  = (const float*)d_in[1];
  const float* b1  = (const float*)d_in[2];
  const float* g1  = (const float*)d_in[3];
  const float* be1 = (const float*)d_in[4];
  const float* m1  = (const float*)d_in[5];
  const float* v1  = (const float*)d_in[6];
  const float* w2  = (const float*)d_in[7];
  const float* b2  = (const float*)d_in[8];
  const float* g2  = (const float*)d_in[9];
  const float* be2 = (const float*)d_in[10];
  const float* m2  = (const float*)d_in[11];
  const float* v2  = (const float*)d_in[12];
  const float* fw1 = (const float*)d_in[13];
  const float* fb1 = (const float*)d_in[14];
  const float* fw2 = (const float*)d_in[15];
  const float* fb2 = (const float*)d_in[16];
  float* out = (float*)d_out;

  char* ws = (char*)d_ws;
  _Float16* w1q  = (_Float16*)(ws);                                   // 32*32
  _Float16* w2q  = (_Float16*)(ws + (64 << 10));                      // 64*288
  _Float16* fw2q = (_Float16*)(ws + (192 << 10));                     // 16*256
  _Float16* fw1q = (_Float16*)(ws + (1 << 20));                       // 256*4096
  _Float16* act1 = (_Float16*)(ws + ((size_t)4 << 20));               // 64 MB
  _Float16* act2 = act1;                                              // alias
  _Float16* pooled1 = (_Float16*)(ws + ((size_t)69 << 20));           // 16 MB
  _Float16* pooled2 = pooled1;                                        // alias
  _Float16* y1 = (_Float16*)(ws + ((size_t)86 << 20));                // 0.5 MB

  ternarize_kernel<<<1, 256, 0, stream>>>(w1, 32, 27, w1q, 32, 32);
  ternarize_kernel<<<1, 256, 0, stream>>>(w2, 64, 288, w2q, 64, 288);
  ternarize_kernel<<<1, 256, 0, stream>>>(fw1, 256, 4096, fw1q, 256, 4096);
  ternarize_kernel<<<1, 256, 0, stream>>>(fw2, 10, 256, fw2q, 16, 256);

  conv1_wmma_kernel<<<8192, 256, 0, stream>>>(x, w1q, b1, g1, be1, m1, v1, act1);

  maxpool_kernel<<<(1024 * 32 * 16 * 16) / 256, 256, 0, stream>>>(
      act1, pooled1, 1024 * 32 * 16 * 16, 32, 32);

  // one block per (image, row-half): 8 waves share the LDS-staged image
  conv2_wmma_kernel<<<2048, 256, 0, stream>>>(pooled1, w2q, b2, g2, be2, m2, v2, act2);

  maxpool_kernel<<<(1024 * 64 * 8 * 8) / 256, 256, 0, stream>>>(
      act2, pooled2, 1024 * 64 * 8 * 8, 16, 16);

  fc1_wmma_kernel<<<128, 256, 0, stream>>>(pooled2, fw1q, fb1, y1);

  fc2_wmma_kernel<<<8, 256, 0, stream>>>(y1, fw2q, fb2, out);
}
